// DeformLVI_61950608277722
// MI455X (gfx1250) — compile-verified, matrix operand order checked by reading, not run on hardware
//
#include <hip/hip_runtime.h>
#include <hip/hip_bf16.h>
#include <cstddef>
#include <cstdint>

// ---------------- problem constants ----------------
#define D_MODEL 256
#define N_HEADS 8
#define N_LEVELS 3
#define N_POINTS 4
#define HEAD_DIM 32
#define D_FF 1024
#define BATCH 16
#define LQ 64
#define S_TOTAL 12096            // 96*96 + 48*48 + 24*24
#define BQ (BATCH * LQ)          // 1024
#define BS ((size_t)BATCH * S_TOTAL) // 193536
#define NCOMBO (N_HEADS * N_LEVELS * N_POINTS) // 96

// ---------------- toolchain feature detection ----------------
#if defined(__has_builtin)
#  if __has_builtin(__builtin_amdgcn_tensor_load_to_lds) && __has_builtin(__builtin_amdgcn_s_wait_tensorcnt)
#    define HAVE_TDM 1
#  else
#    define HAVE_TDM 0
#  endif
#else
#  define HAVE_TDM 0
#endif

#if defined(__has_include)
#  if __has_include(<hip/amd_detail/amd_gfx1250_TDM.h>)
#    define TDM_SIX_ARGS 1      // therock / clang-23 lane
#  else
#    define TDM_SIX_ARGS 0      // ROCm 7.2 / clang-22 lane
#  endif
#else
#  define TDM_SIX_ARGS 0
#endif

// ---------------- vector types ----------------
typedef __attribute__((ext_vector_type(16))) __bf16         bf16x16;
typedef __attribute__((ext_vector_type(8)))  float          floatx8;
typedef __attribute__((ext_vector_type(8)))  unsigned short ushort8;
typedef __attribute__((ext_vector_type(4)))  unsigned int   uint32x4;
typedef __attribute__((ext_vector_type(8)))  int            int32x8;
typedef __attribute__((ext_vector_type(4)))  int            int32x4;

union BF16Frag { ushort8 u[2]; bf16x16 b; };

__device__ __forceinline__ unsigned short f32_to_bf16(float f) {
  unsigned int u = __float_as_uint(f);
  unsigned int r = u + 0x7FFFu + ((u >> 16) & 1u);   // round-to-nearest-even
  return (unsigned short)(r >> 16);
}

#if HAVE_TDM
// Build a 2D Tensor-DMA descriptor (ISA cdna5 §8) and issue TENSOR_LOAD_TO_LDS.
//   tile_d0 elements contiguous (X), tile_d1 rows (Y), row stride = stride0 elems.
//   global_addr points at the TILE start; tensor dims sized so no OOB triggers.
__device__ __forceinline__ void tdm_load_2d(const void* gptr, unsigned lds_off,
                                            unsigned tile_d0, unsigned tile_d1,
                                            unsigned tensor_d0, unsigned tensor_d1,
                                            unsigned stride0_elems)
{
  unsigned long long ga = (unsigned long long)(uintptr_t)gptr;
  uint32x4 g0;
  g0[0] = 1u;                                   // count=1 (valid), user mode
  g0[1] = lds_off;                              // lds_addr (bytes)
  g0[2] = (unsigned)(ga & 0xFFFFFFFFull);       // global_addr[31:0] -> bits 95:64
  g0[3] = (unsigned)((ga >> 32) & 0x01FFFFFFull) | (2u << 30); // addr[56:32] | type=2

  int32x8 g1;
  g1[0] = (int)(2u << 16);                      // wg_mask=0, data_size=2 (4 bytes)
  g1[1] = (int)((tensor_d0 & 0xFFFFu) << 16);   // abar=0 | tensor_dim0[15:0]
  g1[2] = (int)((tensor_d0 >> 16) | ((tensor_d1 & 0xFFFFu) << 16));
  g1[3] = (int)((tensor_d1 >> 16) | (tile_d0 << 16));   // dim1 hi | tile_dim0
  g1[4] = (int)(tile_d1 & 0xFFFFu);             // tile_dim1 | tile_dim2=0
  g1[5] = (int)stride0_elems;                   // tensor_dim0_stride[31:0]
  g1[6] = 0;                                    // stride0 hi | stride1 lo
  g1[7] = 0;                                    // stride1 hi

  int32x4 z4 = {0, 0, 0, 0};
#if TDM_SIX_ARGS
  int32x8 z8 = {0, 0, 0, 0, 0, 0, 0, 0};
  __builtin_amdgcn_tensor_load_to_lds(g0, g1, z4, z4, z8, 0);
#else
  __builtin_amdgcn_tensor_load_to_lds(g0, g1, z4, z4, 0);
#endif
}
#endif // HAVE_TDM

// =====================================================================
// Generic bf16-WMMA GEMM:  C[M,N] = act(A[M,K] @ B[K,N] + bias[N]) ,
// optional per-row zeroing mask. BM=128, BN=64, BK=32; 256 thr = 8 waves.
// A/B tiles are DMAed into f32 LDS staging by the Tensor Data Mover
// (double buffered, overlapped with convert+WMMA), converted to bf16
// tiles (B transposed), then consumed as 16x16x32 WMMA fragments.
// =====================================================================
#define TILE_M 128
#define TILE_N 64
#define TILE_K 32

__global__ __launch_bounds__(256)
void gemm_bf16_wmma_kernel(const float* __restrict__ A, const float* __restrict__ B,
                           const float* __restrict__ bias, float* __restrict__ C,
                           int M, int N, int K, int relu,
                           const unsigned char* __restrict__ mask)
{
#if HAVE_TDM
  __shared__ float Afs[2][TILE_M][TILE_K];        // 2 x 16 KB f32 staging
  __shared__ float Bfs[2][TILE_K][TILE_N];        // 2 x  8 KB f32 staging
#endif
  __shared__ unsigned short As[TILE_M][TILE_K];   // 8 KB, bf16 bits
  __shared__ unsigned short Bs[TILE_N][TILE_K];   // 4 KB, B transposed

  const int t    = threadIdx.x;
  const int wave = t >> 5;          // 0..7 (wave32)
  const int lane = t & 31;
  const int g    = lane >> 4;       // lane group 0/1
  const int mr   = lane & 15;       // row/col within 16-tile
  const int m0   = blockIdx.x * TILE_M;
  const int n0   = blockIdx.y * TILE_N;

  floatx8 acc[4] = {};
  const int nchunks = K / TILE_K;

#if HAVE_TDM
  // prime the pipeline: chunk 0 into buffer 0 (one TDM issue per block: wave 0)
  if (wave == 0) {
    tdm_load_2d(A + (size_t)m0 * K, (unsigned)(uintptr_t)&Afs[0][0][0],
                TILE_K, TILE_M, (unsigned)K, (unsigned)M, (unsigned)K);
    tdm_load_2d(B + n0, (unsigned)(uintptr_t)&Bfs[0][0][0],
                TILE_N, TILE_K, (unsigned)N, (unsigned)K, (unsigned)N);
  }
#endif

  for (int c = 0; c < nchunks; ++c) {
    const int kc  = c * TILE_K;
    const int cur = c & 1;

#if HAVE_TDM
    if (wave == 0) __builtin_amdgcn_s_wait_tensorcnt(0);  // chunk c landed in LDS
    __syncthreads();
    if (wave == 0 && c + 1 < nchunks) {                   // overlap next DMA
      const int kn = kc + TILE_K;
      tdm_load_2d(A + (size_t)m0 * K + kn, (unsigned)(uintptr_t)&Afs[cur ^ 1][0][0],
                  TILE_K, TILE_M, (unsigned)K, (unsigned)M, (unsigned)K);
      tdm_load_2d(B + (size_t)kn * N + n0, (unsigned)(uintptr_t)&Bfs[cur ^ 1][0][0],
                  TILE_N, TILE_K, (unsigned)N, (unsigned)K, (unsigned)N);
    }
    // convert f32 staging -> bf16 WMMA tiles (B transposed)
#pragma unroll
    for (int j = 0; j < 16; ++j) {
      int idx = t + j * 256;
      int row = idx >> 5, kk = idx & 31;
      As[row][kk] = f32_to_bf16(Afs[cur][row][kk]);
    }
#pragma unroll
    for (int j = 0; j < 8; ++j) {
      int idx = t + j * 256;
      int kr = idx >> 6, nn = idx & 63;
      Bs[nn][kr] = f32_to_bf16(Bfs[cur][kr][nn]);
    }
#else
    // fallback staging: direct coalesced global loads + convert
#pragma unroll
    for (int i = 0; i < 4; ++i) {
      int q   = t + i * 256;
      int row = q >> 3;
      int kq  = (q & 7) << 2;
      const float4 v = *(const float4*)(A + (size_t)(m0 + row) * K + kc + kq);
      As[row][kq + 0] = f32_to_bf16(v.x);
      As[row][kq + 1] = f32_to_bf16(v.y);
      As[row][kq + 2] = f32_to_bf16(v.z);
      As[row][kq + 3] = f32_to_bf16(v.w);
    }
#pragma unroll
    for (int i = 0; i < 2; ++i) {
      int q  = t + i * 256;
      int kr = q >> 4;
      int nq = (q & 15) << 2;
      const float4 v = *(const float4*)(B + (size_t)(kc + kr) * N + n0 + nq);
      Bs[nq + 0][kr] = f32_to_bf16(v.x);
      Bs[nq + 1][kr] = f32_to_bf16(v.y);
      Bs[nq + 2][kr] = f32_to_bf16(v.z);
      Bs[nq + 3][kr] = f32_to_bf16(v.w);
    }
#endif
    __syncthreads();

    // ---- A fragment (16x32 bf16): lane L%16 = row, K-halves per ISA layout ----
    BF16Frag af;
    af.u[0] = *(const ushort8*)(&As[wave * 16 + mr][8 * g]);        // K = 8g..8g+7
    af.u[1] = *(const ushort8*)(&As[wave * 16 + mr][16 + 8 * g]);   // K = 16+8g..+7

#pragma unroll
    for (int nt = 0; nt < 4; ++nt) {
      BF16Frag bf;
      bf.u[0] = *(const ushort8*)(&Bs[nt * 16 + mr][8 * g]);
      bf.u[1] = *(const ushort8*)(&Bs[nt * 16 + mr][16 + 8 * g]);
      acc[nt] = __builtin_amdgcn_wmma_f32_16x16x32_bf16(
          false, af.b, false, bf.b, (short)0, acc[nt], false, false);
    }
    __syncthreads();
  }

  // ---- epilogue: C layout lane L reg gg -> row = gg + 8*(L/16), col = L%16 ----
#pragma unroll
  for (int nt = 0; nt < 4; ++nt) {
    int col  = n0 + nt * 16 + mr;
    float bv = bias ? bias[col] : 0.f;
#pragma unroll
    for (int gg = 0; gg < 8; ++gg) {
      int row = m0 + wave * 16 + 8 * g + gg;
      float v = acc[nt][gg] + bv;
      if (relu) v = fmaxf(v, 0.f);
      if (mask && mask[row]) v = 0.f;
      C[(size_t)row * N + col] = v;
    }
  }
}

// =====================================================================
// Small projections: ref (sigmoid), offsets, attention weights (softmax),
// fused into sampling locations + weights. One block per (b, q).
// =====================================================================
__global__ __launch_bounds__(256)
void small_proj_kernel(const float* __restrict__ text,
                       const float* __restrict__ ref_w, const float* __restrict__ ref_b,
                       const float* __restrict__ off_w, const float* __restrict__ off_b,
                       const float* __restrict__ aw_w,  const float* __restrict__ aw_b,
                       const int* __restrict__ shapes,
                       float* __restrict__ loc, float* __restrict__ wout)
{
  __shared__ float x[D_MODEL];
  __shared__ float offb[NCOMBO * 2];
  __shared__ float awb[NCOMBO];
  __shared__ float refb[N_LEVELS * 2];

  const int bq = blockIdx.x;
  const int t  = threadIdx.x;
  x[t] = text[(size_t)bq * D_MODEL + t];
  __syncthreads();

  if (t < NCOMBO * 2) {                 // 192 offset columns
    float s = off_b[t];
    for (int k = 0; k < D_MODEL; ++k) s += x[k] * off_w[k * (NCOMBO * 2) + t];
    offb[t] = s;
  }
  if (t < NCOMBO) {                     // 96 attention-weight columns
    float s = aw_b[t];
    for (int k = 0; k < D_MODEL; ++k) s += x[k] * aw_w[k * NCOMBO + t];
    awb[t] = s;
  }
  if (t >= 96 && t < 96 + N_LEVELS * 2) {   // 6 reference columns
    int n = t - 96;
    float s = ref_b[n];
    for (int k = 0; k < D_MODEL; ++k) s += x[k] * ref_w[k * (N_LEVELS * 2) + n];
    refb[n] = 1.f / (1.f + __expf(-s));     // sigmoid
  }
  __syncthreads();

  if (t < N_HEADS) {                    // softmax over L*P = 12 per head
    float mx = -1e30f;
    for (int j = 0; j < 12; ++j) mx = fmaxf(mx, awb[t * 12 + j]);
    float e[12], sum = 0.f;
    for (int j = 0; j < 12; ++j) { e[j] = __expf(awb[t * 12 + j] - mx); sum += e[j]; }
    float inv = 1.f / sum;
    for (int j = 0; j < 12; ++j) awb[t * 12 + j] = e[j] * inv;
  }
  __syncthreads();

  if (t < NCOMBO) {                     // combo t = h*12 + l*4 + p
    int l  = (t % 12) >> 2;
    int Hl = shapes[l * 2 + 0];
    int Wl = shapes[l * 2 + 1];
    float lx = refb[l * 2 + 0] + offb[t * 2 + 0] / (float)Wl;  // normalizer=(W,H)
    float ly = refb[l * 2 + 1] + offb[t * 2 + 1] / (float)Hl;
    loc[((size_t)bq * NCOMBO + t) * 2 + 0] = lx;
    loc[((size_t)bq * NCOMBO + t) * 2 + 1] = ly;
    wout[(size_t)bq * NCOMBO + t] = awb[t];
  }
}

// =====================================================================
// Deformable bilinear sampling. One block per (b,q); thread = (head, dim)
// so the 32 lanes of each head gather fully-coalesced 128B bursts.
// =====================================================================
__global__ __launch_bounds__(256)
void deform_sample_kernel(const float* __restrict__ value,
                          const float* __restrict__ loc, const float* __restrict__ w,
                          const int* __restrict__ shapes, const int* __restrict__ lstart,
                          float* __restrict__ ctx)
{
  __shared__ float slx[NCOMBO], sly[NCOMBO], sw[NCOMBO];
  const int bq = blockIdx.x;
  const int b  = bq >> 6;              // LQ = 64
  const int t  = threadIdx.x;
  if (t < NCOMBO) {
    slx[t] = loc[((size_t)bq * NCOMBO + t) * 2 + 0];
    sly[t] = loc[((size_t)bq * NCOMBO + t) * 2 + 1];
    sw[t]  = w[(size_t)bq * NCOMBO + t];
  }
  __syncthreads();

  const int h = t >> 5, d = t & 31;
  const float* vb = value + (size_t)b * S_TOTAL * D_MODEL + h * HEAD_DIM + d;
  float acc = 0.f;

  for (int l = 0; l < N_LEVELS; ++l) {
    const int Hl = shapes[l * 2 + 0];
    const int Wl = shapes[l * 2 + 1];
    const int st = lstart[l];
#pragma unroll
    for (int p = 0; p < N_POINTS; ++p) {
      const int c = h * 12 + l * 4 + p;
      float xx = slx[c] * (float)Wl - 0.5f;
      float yy = sly[c] * (float)Hl - 0.5f;
      float x0f = floorf(xx), y0f = floorf(yy);
      float fx = xx - x0f, fy = yy - y0f;
      int ix0 = (int)x0f, iy0 = (int)y0f;
      int ix1 = ix0 + 1,  iy1 = iy0 + 1;
      float v00 = 0.f, v10 = 0.f, v01 = 0.f, v11 = 0.f;
      bool okx0 = (ix0 >= 0) & (ix0 < Wl);
      bool okx1 = (ix1 >= 0) & (ix1 < Wl);
      if (iy0 >= 0 && iy0 < Hl) {
        if (okx0) v00 = vb[(size_t)(st + iy0 * Wl + ix0) * D_MODEL];
        if (okx1) v10 = vb[(size_t)(st + iy0 * Wl + ix1) * D_MODEL];
      }
      if (iy1 >= 0 && iy1 < Hl) {
        if (okx0) v01 = vb[(size_t)(st + iy1 * Wl + ix0) * D_MODEL];
        if (okx1) v11 = vb[(size_t)(st + iy1 * Wl + ix1) * D_MODEL];
      }
      acc += sw[c] * (v00 * (1.f - fx) * (1.f - fy) + v10 * fx * (1.f - fy) +
                      v01 * (1.f - fx) * fy         + v11 * fx * fy);
    }
  }
  ctx[(size_t)bq * D_MODEL + t] = acc;
}

// =====================================================================
// Fused residual-add + LayerNorm: out = LN(a + b) * gamma + beta.
// =====================================================================
__global__ __launch_bounds__(256)
void add_ln_kernel(const float* __restrict__ a, const float* __restrict__ bsrc,
                   const float* __restrict__ gamma, const float* __restrict__ beta,
                   float* __restrict__ out)
{
  __shared__ float red[256];
  const int row = blockIdx.x, t = threadIdx.x;
  float x = a[(size_t)row * D_MODEL + t] + bsrc[(size_t)row * D_MODEL + t];
  red[t] = x; __syncthreads();
  for (int s = 128; s > 0; s >>= 1) { if (t < s) red[t] += red[t + s]; __syncthreads(); }
  float mean = red[0] * (1.f / D_MODEL);
  __syncthreads();
  float d = x - mean;
  red[t] = d * d; __syncthreads();
  for (int s = 128; s > 0; s >>= 1) { if (t < s) red[t] += red[t + s]; __syncthreads(); }
  float var = red[0] * (1.f / D_MODEL);
  out[(size_t)row * D_MODEL + t] = d * rsqrtf(var + 1e-5f) * gamma[t] + beta[t];
}

// =====================================================================
// Host-side launch
// =====================================================================
extern "C" void kernel_launch(void* const* d_in, const int* in_sizes, int n_in,
                              void* d_out, int out_size, void* d_ws, size_t ws_size,
                              hipStream_t stream)
{
  const float* text      = (const float*)d_in[0];
  const float* vis_value = (const float*)d_in[1];
  const int*   shapes    = (const int*)d_in[2];
  const int*   lstart    = (const int*)d_in[3];
  const unsigned char* mask = (const unsigned char*)d_in[4]; // jnp.bool_
  const float* ref_w  = (const float*)d_in[5];
  const float* ref_b  = (const float*)d_in[6];
  const float* off_w  = (const float*)d_in[7];
  const float* off_b  = (const float*)d_in[8];
  const float* aw_w   = (const float*)d_in[9];
  const float* aw_b   = (const float*)d_in[10];
  const float* val_w  = (const float*)d_in[11];
  const float* val_b  = (const float*)d_in[12];
  const float* proj_w = (const float*)d_in[13];
  const float* proj_b = (const float*)d_in[14];
  const float* out_w  = (const float*)d_in[15];
  const float* out_b  = (const float*)d_in[16];
  const float* n1_g   = (const float*)d_in[17];
  const float* n1_b   = (const float*)d_in[18];
  const float* f1_w   = (const float*)d_in[19];
  const float* f1_b   = (const float*)d_in[20];
  const float* f2_w   = (const float*)d_in[21];
  const float* f2_b   = (const float*)d_in[22];
  const float* n2_g   = (const float*)d_in[23];
  const float* n2_b   = (const float*)d_in[24];

  // workspace carve-up (floats)
  float* ws    = (float*)d_ws;
  size_t o     = 0;
  float* value = ws + o; o += BS * D_MODEL;            // 193536 x 256
  float* locb  = ws + o; o += (size_t)BQ * NCOMBO * 2; // sampling locations
  float* wb    = ws + o; o += (size_t)BQ * NCOMBO;     // softmaxed weights
  float* ctx   = ws + o; o += (size_t)BQ * D_MODEL;
  float* tmp1  = ws + o; o += (size_t)BQ * D_MODEL;
  float* tmp2  = ws + o; o += (size_t)BQ * D_MODEL;
  float* tbuf  = ws + o; o += (size_t)BQ * D_MODEL;
  float* h1    = ws + o; o += (size_t)BQ * D_FF;
  (void)ws_size; (void)in_sizes; (void)n_in; (void)out_size;

  // 1) sampling locations + weights from text tokens
  small_proj_kernel<<<BQ, 256, 0, stream>>>(text, ref_w, ref_b, off_w, off_b,
                                            aw_w, aw_b, shapes, locb, wb);

  // 2) value projection (big memory-bound GEMM) with key-padding mask
  {
    dim3 grid((unsigned)(BS / TILE_M), D_MODEL / TILE_N);
    gemm_bf16_wmma_kernel<<<grid, 256, 0, stream>>>(vis_value, val_w, val_b, value,
                                                    (int)BS, D_MODEL, D_MODEL, 0, mask);
  }

  // 3) deformable bilinear sampling -> ctx
  deform_sample_kernel<<<BQ, 256, 0, stream>>>(value, locb, wb, shapes, lstart, ctx);

  // 4) ctx @ proj_w + proj_b
  {
    dim3 grid(BQ / TILE_M, D_MODEL / TILE_N);
    gemm_bf16_wmma_kernel<<<grid, 256, 0, stream>>>(ctx, proj_w, proj_b, tmp1,
                                                    BQ, D_MODEL, D_MODEL, 0, nullptr);
  }
  // 5) @ out_w + out_b
  {
    dim3 grid(BQ / TILE_M, D_MODEL / TILE_N);
    gemm_bf16_wmma_kernel<<<grid, 256, 0, stream>>>(tmp1, out_w, out_b, tmp2,
                                                    BQ, D_MODEL, D_MODEL, 0, nullptr);
  }
  // 6) t = LN(text + attn)
  add_ln_kernel<<<BQ, 256, 0, stream>>>(text, tmp2, n1_g, n1_b, tbuf);

  // 7) h1 = relu(t @ f1_w + f1_b)
  {
    dim3 grid(BQ / TILE_M, D_FF / TILE_N);
    gemm_bf16_wmma_kernel<<<grid, 256, 0, stream>>>(tbuf, f1_w, f1_b, h1,
                                                    BQ, D_FF, D_MODEL, 1, nullptr);
  }
  // 8) ffn = h1 @ f2_w + f2_b
  {
    dim3 grid(BQ / TILE_M, D_MODEL / TILE_N);
    gemm_bf16_wmma_kernel<<<grid, 256, 0, stream>>>(h1, f2_w, f2_b, tmp2,
                                                    BQ, D_MODEL, D_FF, 0, nullptr);
  }
  // 9) out = LN(t + ffn)
  add_ln_kernel<<<BQ, 256, 0, stream>>>(tbuf, tmp2, n2_g, n2_b, (float*)d_out);
}
